// CaptionNet_85126251806758
// MI455X (gfx1250) — compile-verified, matrix operand order checked by reading, not run on hardware
//
#include <hip/hip_runtime.h>
#include <hip/hip_bf16.h>
#include <math.h>
#include <stdint.h>

// Problem constants (from reference)
#define B_   128
#define C_   512
#define P_   196
#define T_   32
#define V_   10403
#define E_   128
#define U_   256
#define KXW  896        // E + C + U  (shared K for gates/logits GEMMs)
#define G4U  1024       // 4*U
#define VPAD 10432      // W_out rows padded to 163*64

typedef __attribute__((ext_vector_type(8)))  __bf16 v8bf;
typedef __attribute__((ext_vector_type(16))) __bf16 v16bf;
typedef __attribute__((ext_vector_type(8)))  float  v8f;

// ---------------------------------------------------------------------------
// Load a 16-element bf16 fragment: two contiguous 16B chunks at p and p+16
// (CDNA5 16-bit A/B fragment layout: a lane holds K chunks {c, c+16}).
// ---------------------------------------------------------------------------
__device__ __forceinline__ v16bf ld_frag(const __bf16* p) {
    v8bf lo = *(const v8bf*)(p);
    v8bf hi = *(const v8bf*)(p + 16);
    v16bf r;
#pragma unroll
    for (int i = 0; i < 8; ++i) { r[i] = lo[i]; r[i + 8] = hi[i]; }
    return r;
}

// ---------------------------------------------------------------------------
// Generic D[M,N] = A[M,K] * W[N,K]^T + bias, bf16 in / f32 out.
// One wave (32 threads) per block computes a (16*MT)x64 strip:
//   grid.x = N/64 strips, grid.y = M/(16*MT) tiles.
// Per K-step: MT A-fragments + 4 B-fragments -> 4*MT WMMAs.
// Rows are row-major, 16B aligned (K multiple of 32). Stores cols < Nvalid.
// ---------------------------------------------------------------------------
template<int MT>
__global__ __launch_bounds__(32)
void wmma_gemm_xwt(const __bf16* __restrict__ A, int lda,
                   const __bf16* __restrict__ W, int ldw,
                   const float* __restrict__ bias, int Nvalid,
                   float* __restrict__ D, long ldd, int K)
{
    const int lane  = threadIdx.x & 31;
    const int mBase = blockIdx.y * (16 * MT);
    const int nBase = blockIdx.x << 6;
    const int half  = lane >> 4;      // which K-chunk pair this lane holds
    const int r16   = lane & 15;

    const __bf16* aRow[MT];
#pragma unroll
    for (int m = 0; m < MT; ++m)
        aRow[m] = A + (long)(mBase + 16 * m + r16) * lda + half * 8;

    const __bf16* wRow0 = W + (long)(nBase + r16) * ldw + half * 8;
    const __bf16* wRow1 = wRow0 + 16L * ldw;
    const __bf16* wRow2 = wRow0 + 32L * ldw;
    const __bf16* wRow3 = wRow0 + 48L * ldw;

    v8f acc[MT][4] = {};

    for (int k = 0; k < K; k += 32) {
        __builtin_prefetch((const void*)(wRow0 + k + 64), 0, 0);
        __builtin_prefetch((const void*)(aRow[0] + k + 64), 0, 0);

        v16bf a[MT];
#pragma unroll
        for (int m = 0; m < MT; ++m) a[m] = ld_frag(aRow[m] + k);

        v16bf b0 = ld_frag(wRow0 + k);
        v16bf b1 = ld_frag(wRow1 + k);
        v16bf b2 = ld_frag(wRow2 + k);
        v16bf b3 = ld_frag(wRow3 + k);

#pragma unroll
        for (int m = 0; m < MT; ++m) {
            acc[m][0] = __builtin_amdgcn_wmma_f32_16x16x32_bf16(false, a[m], false, b0, (short)0, acc[m][0], false, false);
            acc[m][1] = __builtin_amdgcn_wmma_f32_16x16x32_bf16(false, a[m], false, b1, (short)0, acc[m][1], false, false);
            acc[m][2] = __builtin_amdgcn_wmma_f32_16x16x32_bf16(false, a[m], false, b2, (short)0, acc[m][2], false, false);
            acc[m][3] = __builtin_amdgcn_wmma_f32_16x16x32_bf16(false, a[m], false, b3, (short)0, acc[m][3], false, false);
        }
    }

    // C/D layout: lane (l&15) = column, VGPR r = row r (+8 for high lanes)
#pragma unroll
    for (int m = 0; m < MT; ++m) {
        const int row0 = mBase + 16 * m + half * 8;
#pragma unroll
        for (int j = 0; j < 4; ++j) {
            const int col = nBase + j * 16 + r16;
            if (col < Nvalid) {
                const float bv = bias ? bias[col] : 0.0f;
#pragma unroll
                for (int r = 0; r < 8; ++r)
                    D[(long)(row0 + r) * ldd + col] = acc[m][j][r] + bv;
            }
        }
    }
}

// ---------------------------------------------------------------------------
// Preprocessing kernels
// ---------------------------------------------------------------------------

// values_bf[b,p,c] = bf16(feat[b,c,p]); mean[b,c] = mean over p
__global__ void k_values_mean(const float* __restrict__ feat,
                              __bf16* __restrict__ values_bf,
                              float* __restrict__ mean)
{
    int idx = blockIdx.x * blockDim.x + threadIdx.x;   // B*C threads
    if (idx >= B_ * C_) return;
    int b = idx / C_, c = idx % C_;
    const float* src = feat + (long)b * C_ * P_ + (long)c * P_;
    float s = 0.f;
    for (int p = 0; p < P_; ++p) {
        float v = src[p];
        s += v;
        values_bf[((long)b * P_ + p) * C_ + c] = (__bf16)v;
    }
    mean[idx] = s * (1.0f / (float)P_);
}

// Wcomb[row, 0:640] = W_ih ; Wcomb[row, 640:896] = W_hh ; biasg = b_ih + b_hh
__global__ void k_wcomb(const float* __restrict__ W_ih, const float* __restrict__ W_hh,
                        const float* __restrict__ b_ih, const float* __restrict__ b_hh,
                        __bf16* __restrict__ Wcomb, float* __restrict__ biasg)
{
    int i = blockIdx.x * blockDim.x + threadIdx.x;     // G4U*KXW threads
    if (i >= G4U * KXW) return;
    int row = i / KXW, col = i % KXW;
    float v = (col < (E_ + C_)) ? W_ih[(long)row * (E_ + C_) + col]
                                : W_hh[(long)row * U_ + (col - (E_ + C_))];
    Wcomb[i] = (__bf16)v;
    if (i < G4U) biasg[i] = b_ih[i] + b_hh[i];
}

__global__ void k_wout(const float* __restrict__ W_out, __bf16* __restrict__ Wout_bf)
{
    int i = blockIdx.x * blockDim.x + threadIdx.x;     // VPAD*KXW threads
    if (i >= VPAD * KXW) return;
    int row = i / KXW, col = i % KXW;
    Wout_bf[i] = (__bf16)((row < V_) ? W_out[(long)row * KXW + col] : 0.0f);
}

__global__ void k_wkey(const float* __restrict__ W_key, __bf16* __restrict__ Wkey_bf)
{
    int i = blockIdx.x * blockDim.x + threadIdx.x;     // U*C threads
    if (i >= U_ * C_) return;
    Wkey_bf[i] = (__bf16)W_key[i];
}

// emb_bf[(b*T+t), e] = bf16(emb[captions_ix[b,t], e])
__global__ void k_emb_gather(const float* __restrict__ emb, const int* __restrict__ cap,
                             __bf16* __restrict__ emb_bf)
{
    int i = blockIdx.x * blockDim.x + threadIdx.x;     // B*T*E threads
    if (i >= B_ * T_ * E_) return;
    int bt = i / E_, e = i % E_;
    int tok = cap[bt];
    emb_bf[i] = (__bf16)emb[(long)tok * E_ + e];
}

// h = mean@W_h0^T + b_h0 ; c = q = mean@W_c0^T + b_c0   (tiny f32 GEMV, 1 block/b)
__global__ void k_init_hc(const float* __restrict__ mean,
                          const float* __restrict__ W_h0, const float* __restrict__ b_h0,
                          const float* __restrict__ W_c0, const float* __restrict__ b_c0,
                          float* __restrict__ h, float* __restrict__ c, float* __restrict__ q)
{
    __shared__ float mf[C_];
    int b = blockIdx.x, tid = threadIdx.x;             // 256 threads
    mf[tid]       = mean[b * C_ + tid];
    mf[tid + 256] = mean[b * C_ + 256 + tid];
    __syncthreads();
    float sh = b_h0[tid], sc = b_c0[tid];
    const float* wh = W_h0 + (long)tid * C_;
    const float* wc = W_c0 + (long)tid * C_;
    for (int k = 0; k < C_; ++k) { sh += mf[k] * wh[k]; sc += mf[k] * wc[k]; }
    h[b * U_ + tid] = sh;
    c[b * U_ + tid] = sc;
    q[b * U_ + tid] = sc;
}

// ---------------------------------------------------------------------------
// Per-step attention + activation assembly. One block per batch element.
// Fills x = [emb_t | attn | h_prev] (bf16) and y = [attn | emb_t | ...],
// writes attention map to d_out.
// ---------------------------------------------------------------------------
__global__ void k_attention(const float* __restrict__ q, const float* __restrict__ keys,
                            const float* __restrict__ feat, const __bf16* __restrict__ emb_bf,
                            const float* __restrict__ h,
                            __bf16* __restrict__ x, __bf16* __restrict__ y,
                            float* __restrict__ maps, int t)
{
    __shared__ float qs[U_];
    __shared__ float sc[256];
    __shared__ float red[256];
    const int b = blockIdx.x, tid = threadIdx.x;       // 256 threads
    const float scale = 0.0625f;                       // 1/sqrt(U)

    qs[tid] = q[b * U_ + tid];
    __syncthreads();

    float s = -3.0e38f;
    if (tid < P_) {
        const float* kr = keys + ((long)b * P_ + tid) * U_;
        float d = 0.f;
        for (int u = 0; u < U_; ++u) d += qs[u] * kr[u];
        s = d * scale;
    }
    sc[tid] = s;
    red[tid] = s;
    __syncthreads();
    for (int w = 128; w > 0; w >>= 1) {
        if (tid < w) red[tid] = fmaxf(red[tid], red[tid + w]);
        __syncthreads();
    }
    const float mx = red[0];
    __syncthreads();
    float e = (tid < P_) ? __expf(sc[tid] - mx) : 0.0f;
    red[tid] = e;
    __syncthreads();
    for (int w = 128; w > 0; w >>= 1) {
        if (tid < w) red[tid] += red[tid + w];
        __syncthreads();
    }
    const float inv = 1.0f / red[0];
    __syncthreads();
    const float wgt = e * inv;
    sc[tid] = wgt;
    if (tid < P_) maps[((long)b * T_ + t) * P_ + tid] = wgt;
    __syncthreads();

    // attn[c] = sum_p w[p] * feat[b,c,p]   (p contiguous in memory)
    for (int c = tid; c < C_; c += 256) {
        const float* fr = feat + ((long)b * C_ + c) * P_;
        float a = 0.f;
        for (int p = 0; p < P_; ++p) a += sc[p] * fr[p];
        __bf16 ab = (__bf16)a;
        x[(long)b * KXW + E_ + c] = ab;     // x: [emb|attn|h]
        y[(long)b * KXW + c]      = ab;     // y: [attn|emb|h]
    }
    // emb slots
    if (tid < E_) {
        __bf16 ev = emb_bf[((long)b * T_ + t) * E_ + tid];
        x[(long)b * KXW + tid]            = ev;
        y[(long)b * KXW + C_ + tid]       = ev;
    }
    // h_prev slot of x (gates use previous hidden)
    x[(long)b * KXW + E_ + C_ + tid] = (__bf16)h[b * U_ + tid];
}

// ---------------------------------------------------------------------------
// LSTM cell: consumes gates [B,4U], updates h/c/q, writes h_new slot of y.
// ---------------------------------------------------------------------------
__device__ __forceinline__ float sigmf(float v) { return 1.0f / (1.0f + __expf(-v)); }

__global__ void k_cell(const float* __restrict__ gates,
                       float* __restrict__ h, float* __restrict__ c, float* __restrict__ q,
                       __bf16* __restrict__ y)
{
    int b = blockIdx.x, u = threadIdx.x;               // 256 threads
    const float* g = gates + (long)b * G4U;
    float ig = sigmf(g[u]);
    float fg = sigmf(g[U_ + u]);
    float gg = tanhf(g[2 * U_ + u]);
    float og = sigmf(g[3 * U_ + u]);
    float cn = fg * c[b * U_ + u] + ig * gg;
    float hn = og * tanhf(cn);
    c[b * U_ + u] = cn;
    h[b * U_ + u] = hn;
    q[b * U_ + u] = hn;
    y[(long)b * KXW + C_ + E_ + u] = (__bf16)hn;
}

// ---------------------------------------------------------------------------
// Host launcher
// ---------------------------------------------------------------------------
extern "C" void kernel_launch(void* const* d_in, const int* in_sizes, int n_in,
                              void* d_out, int out_size, void* d_ws, size_t ws_size,
                              hipStream_t stream)
{
    (void)in_sizes; (void)n_in; (void)out_size; (void)ws_size;

    const float* feat   = (const float*)d_in[0];   // [B,C,P]
    const int*   cap    = (const int*)  d_in[1];   // [B,T]
    const float* W_h0   = (const float*)d_in[2];
    const float* b_h0   = (const float*)d_in[3];
    const float* W_c0   = (const float*)d_in[4];
    const float* b_c0   = (const float*)d_in[5];
    const float* W_key  = (const float*)d_in[6];
    const float* b_key  = (const float*)d_in[7];
    const float* emb    = (const float*)d_in[8];
    const float* W_ih   = (const float*)d_in[9];
    const float* b_ih   = (const float*)d_in[10];
    const float* W_hh   = (const float*)d_in[11];
    const float* b_hh   = (const float*)d_in[12];
    const float* W_out  = (const float*)d_in[13];
    const float* b_out  = (const float*)d_in[14];

    float* out_logits = (float*)d_out;                              // [B,T,V]
    float* out_maps   = (float*)d_out + (long)B_ * T_ * V_;         // [B,T,P]

    // workspace carve-up (256B aligned)
    uintptr_t base = (uintptr_t)d_ws;
    auto take = [&](size_t bytes) -> void* {
        uintptr_t p = (base + 255) & ~(uintptr_t)255;
        base = p + bytes;
        return (void*)p;
    };
    __bf16* values_bf = (__bf16*)take((size_t)B_ * P_ * C_ * 2);    // 25.7 MB
    __bf16* Wout_bf   = (__bf16*)take((size_t)VPAD * KXW * 2);      // 18.7 MB
    __bf16* Wcomb_bf  = (__bf16*)take((size_t)G4U * KXW * 2);
    __bf16* Wkey_bf   = (__bf16*)take((size_t)U_ * C_ * 2);
    __bf16* emb_bf    = (__bf16*)take((size_t)B_ * T_ * E_ * 2);
    float*  keys      = (float*) take((size_t)B_ * P_ * U_ * 4);    // 25.7 MB
    float*  mean      = (float*) take((size_t)B_ * C_ * 4);
    float*  h         = (float*) take((size_t)B_ * U_ * 4);
    float*  c         = (float*) take((size_t)B_ * U_ * 4);
    float*  q         = (float*) take((size_t)B_ * U_ * 4);
    float*  gates     = (float*) take((size_t)B_ * G4U * 4);
    float*  biasg     = (float*) take((size_t)G4U * 4);
    __bf16* x         = (__bf16*)take((size_t)B_ * KXW * 2);
    __bf16* y         = (__bf16*)take((size_t)B_ * KXW * 2);

    // ---- one-time preprocessing ----
    k_values_mean<<<(B_ * C_ + 255) / 256, 256, 0, stream>>>(feat, values_bf, mean);
    k_wcomb<<<(G4U * KXW + 255) / 256, 256, 0, stream>>>(W_ih, W_hh, b_ih, b_hh, Wcomb_bf, biasg);
    k_wout<<<((size_t)VPAD * KXW + 255) / 256, 256, 0, stream>>>(W_out, Wout_bf);
    k_wkey<<<(U_ * C_ + 255) / 256, 256, 0, stream>>>(W_key, Wkey_bf);
    k_emb_gather<<<(B_ * T_ * E_ + 255) / 256, 256, 0, stream>>>(emb, cap, emb_bf);
    k_init_hc<<<B_, 256, 0, stream>>>(mean, W_h0, b_h0, W_c0, b_c0, h, c, q);

    // keys[B*P, U] = values[B*P, C] x W_key[U, C]^T + b_key   (32-row strips)
    wmma_gemm_xwt<2><<<dim3(U_ / 64, (B_ * P_) / 32), 32, 0, stream>>>(
        values_bf, C_, Wkey_bf, C_, b_key, U_, keys, (long)U_, C_);

    // ---- sequential decode ----
    for (int t = 0; t < T_; ++t) {
        k_attention<<<B_, 256, 0, stream>>>(q, keys, feat, emb_bf, h, x, y, out_maps, t);

        // gates[B, 4U] = x[B, 896] x Wcomb[1024, 896]^T + (b_ih + b_hh)
        // 16-row strips: keep 128 independent waves on this tiny GEMM
        wmma_gemm_xwt<1><<<dim3(G4U / 64, B_ / 16), 32, 0, stream>>>(
            x, KXW, Wcomb_bf, KXW, biasg, G4U, gates, (long)G4U, KXW);

        k_cell<<<B_, 256, 0, stream>>>(gates, h, c, q, y);

        // logits[B, V] = y[B, 896] x Wout[VPAD, 896]^T + b_out  -> d_out[:, t, :]
        // 32-row strips: 8 WMMA per 6 fragment loads, 652 waves
        wmma_gemm_xwt<2><<<dim3(VPAD / 64, B_ / 32), 32, 0, stream>>>(
            y, KXW, Wout_bf, KXW, b_out, V_, out_logits + (long)t * V_, (long)T_ * V_, KXW);
    }
}
